// MultiHeadAttention_26474178413105
// MI455X (gfx1250) — matrix operand order repaired and checked by reading, not compiled, not run on hardware
//
#include <hip/hip_runtime.h>
#include <hip/hip_bf16.h>

// ---------------------------------------------------------------------------
// Multi-head attention for gfx1250 (MI455X), all GEMM stages on
// v_wmma_f32_16x16x32_f16 (f16 inputs, f32 accumulate), flash-attention
// style online softmax so the S x S score matrix is never materialized.
// ---------------------------------------------------------------------------

typedef _Float16 h16;
typedef __attribute__((ext_vector_type(16))) _Float16 v16h;
typedef __attribute__((ext_vector_type(8)))  _Float16 v8h;
typedef __attribute__((ext_vector_type(8)))  float    v8f;

constexpr int kB = 2;
constexpr int kS = 2048;
constexpr int kD = 1024;
constexpr int kH = 16;
constexpr int kE = 64;

constexpr float kScale = 0.125f;          // 1/sqrt(64)
constexpr float kLog2e = 1.44269504088896f;

// ---- WMMA wrapper ---------------------------------------------------------
__device__ __forceinline__ v8f wmma_f16(v16h a, v16h b, v8f c) {
  // 8 args: (neg_a, A, neg_b, B, c_mod, C, reuse_a, reuse_b)
  return __builtin_amdgcn_wmma_f32_16x16x32_f16(false, a, false, b,
                                                (short)0, c, false, false);
}

// A-fragment (16x32, f16, row-major source with leading dim `ld`):
// lane(m = lane&15, half = lane>>4) holds k in [k0+16h, k0+16h+16)
// => one contiguous 32B load.
__device__ __forceinline__ v16h loadA(const h16* tile, int ld, int k0,
                                      int m, int half) {
  return *(const v16h*)(tile + (size_t)m * ld + k0 + 16 * half);
}

// B-fragment (32x16): lane holds 16 consecutive K values for its column;
// caller passes a pointer to K-contiguous storage already offset to
// (column n, k0 + 16*half).
__device__ __forceinline__ v16h loadB(const h16* p) {
  return *(const v16h*)p;
}

__device__ __forceinline__ v8f zero8() {
  v8f z;
#pragma unroll
  for (int i = 0; i < 8; ++i) z[i] = 0.0f;
  return z;
}

// ---- Kernel 1: cast activations f32 -> f16 --------------------------------
__global__ void cast_acts_kernel(const float* __restrict__ q,
                                 const float* __restrict__ v,
                                 h16* __restrict__ q16,
                                 h16* __restrict__ v16, int n) {
  for (int i = blockIdx.x * blockDim.x + threadIdx.x; i < n;
       i += gridDim.x * blockDim.x) {
    q16[i] = (h16)q[i];
    v16[i] = (h16)v[i];
  }
}

// ---- Kernel 2: weight preparation (cast + transpose to K-contiguous) ------
__global__ void prep_weights_kernel(const float* __restrict__ Wq,
                                    const float* __restrict__ Wk,
                                    const float* __restrict__ Wv,
                                    const float* __restrict__ Wo,
                                    h16* __restrict__ Wqt,
                                    h16* __restrict__ Wkt,
                                    h16* __restrict__ Wvt,
                                    h16* __restrict__ Wot) {
  const int stride = gridDim.x * blockDim.x;
  const int nw = kH * kD * kE;          // per-head weights [H,D,E] -> [H,E,D]
  for (int i = blockIdx.x * blockDim.x + threadIdx.x; i < nw; i += stride) {
    int h = i / (kD * kE);
    int r = i % (kD * kE);
    int d = r / kE;
    int e = r % kE;
    size_t dst = ((size_t)h * kE + e) * kD + d;
    Wqt[dst] = (h16)Wq[i];
    Wkt[dst] = (h16)Wk[i];
    Wvt[dst] = (h16)Wv[i];
  }
  const int no = kD * kD;               // Wo [Din,Dout] -> [Dout,Din]
  for (int i = blockIdx.x * blockDim.x + threadIdx.x; i < no; i += stride) {
    int din = i / kD;
    int dout = i % kD;
    Wot[(size_t)dout * kD + din] = (h16)Wo[i];
  }
}

// ---- Kernel 3: fused Q/K/V projections ------------------------------------
// One wave computes one 16x64 tile of Q, K, or V for one (b,h).
// Q,K stored [B,H,S,E]; V stored transposed [B,H,E,S] (K-contiguous for PV).
__global__ void qkv_proj_kernel(const h16* __restrict__ q16,
                                const h16* __restrict__ v16,
                                const h16* __restrict__ Wqt,
                                const h16* __restrict__ Wkt,
                                const h16* __restrict__ Wvt,
                                const float* __restrict__ bq,
                                const float* __restrict__ bk,
                                const float* __restrict__ bv,
                                h16* __restrict__ Qb,
                                h16* __restrict__ Kb,
                                h16* __restrict__ Vt) {
  const int lane = threadIdx.x & 31;
  const int wid = blockIdx.x * (blockDim.x >> 5) + (threadIdx.x >> 5);
  const int st = wid & 127;             // S/16 = 128 row tiles
  const int h  = (wid >> 7) & 15;
  const int b  = (wid >> 11) & 1;
  const int t  = wid >> 12;             // 0=Q, 1=K, 2=V
  const int n = lane & 15, m = n, half = lane >> 4;

  const h16* act = (t == 0 ? q16 : v16) + ((size_t)b * kS + st * 16) * kD;
  const h16* Wt  = (t == 0 ? Wqt : (t == 1 ? Wkt : Wvt)) + (size_t)h * kE * kD;
  const float* bias = (t == 0 ? bq : (t == 1 ? bk : bv)) + h * kE;

  v8f acc[4];
#pragma unroll
  for (int j = 0; j < 4; ++j) acc[j] = zero8();

  for (int kc = 0; kc < kD / 32; ++kc) {
    v16h a = loadA(act, kD, kc * 32, m, half);
#pragma unroll
    for (int j = 0; j < 4; ++j) {
      v16h bf = loadB(Wt + (size_t)(16 * j + n) * kD + kc * 32 + 16 * half);
      acc[j] = wmma_f16(a, bf, acc[j]);
    }
  }

#pragma unroll
  for (int j = 0; j < 4; ++j) {
    const float bi = bias[16 * j + n];
#pragma unroll
    for (int r = 0; r < 8; ++r) {
      const int s = st * 16 + 8 * half + r;
      const float val = acc[j][r] + bi;
      if (t == 2) {
        Vt[((size_t)(b * kH + h) * kE + 16 * j + n) * kS + s] = (h16)val;
      } else {
        h16* dst = (t == 0 ? Qb : Kb);
        dst[((size_t)(b * kH + h) * kS + s) * kE + 16 * j + n] = (h16)val;
      }
    }
  }
}

// ---- Kernel 4: flash attention core ---------------------------------------
// One wave owns one 16-row query tile; streams K/V in chunks of 32 columns
// with online softmax. P tile bounces through LDS to re-shape the score
// D-fragment into an A-fragment for the PV WMMA.
__global__ void flash_attn_kernel(const h16* __restrict__ Qb,
                                  const h16* __restrict__ Kb,
                                  const h16* __restrict__ Vt,
                                  h16* __restrict__ heads) {
  __shared__ alignas(32) h16 lds[4 * 16 * 32];   // 1KB per wave
  const int lane = threadIdx.x & 31;
  const int wv = threadIdx.x >> 5;
  const int wid = blockIdx.x * 4 + wv;
  const int qt = wid & 127;
  const int h  = (wid >> 7) & 15;
  const int b  = wid >> 11;
  const int n = lane & 15, m = n, half = lane >> 4;
  h16* P = &lds[wv * 512];

  const h16* Qtile = Qb + ((size_t)(b * kH + h) * kS + qt * 16) * kE;
  const h16* Kbh   = Kb + (size_t)(b * kH + h) * kS * kE;
  const h16* Vbh   = Vt + (size_t)(b * kH + h) * kE * kS;

  const v16h qa0 = loadA(Qtile, kE, 0,  m, half);   // e = 0..31
  const v16h qa1 = loadA(Qtile, kE, 32, m, half);   // e = 32..63

  v8f acc[4];
#pragma unroll
  for (int j = 0; j < 4; ++j) acc[j] = zero8();
  float rm[8], rs[8];
#pragma unroll
  for (int r = 0; r < 8; ++r) { rm[r] = -1e30f; rs[r] = 0.0f; }

  for (int kc = 0; kc < kS / 32; ++kc) {
    const int sk0 = kc * 32;
    // scores: two 16x16 tiles (k-cols [0,16) and [16,32) of the chunk)
    v8f s0 = zero8(), s1 = zero8();
    {
      v16h k00 = loadB(Kbh + (size_t)(sk0 + n) * kE + 16 * half);
      v16h k10 = loadB(Kbh + (size_t)(sk0 + 16 + n) * kE + 16 * half);
      s0 = wmma_f16(qa0, k00, s0);
      s1 = wmma_f16(qa0, k10, s1);
      v16h k01 = loadB(Kbh + (size_t)(sk0 + n) * kE + 32 + 16 * half);
      v16h k11 = loadB(Kbh + (size_t)(sk0 + 16 + n) * kE + 32 + 16 * half);
      s0 = wmma_f16(qa1, k01, s0);
      s1 = wmma_f16(qa1, k11, s1);
    }
    // online softmax per row (row m=8*half+r lives in the 16 lanes of a half)
#pragma unroll
    for (int r = 0; r < 8; ++r) {
      float a = s0[r] * kScale;
      float c = s1[r] * kScale;
      float t = fmaxf(a, c);
      t = fmaxf(t, __shfl_xor(t, 1, 32));
      t = fmaxf(t, __shfl_xor(t, 2, 32));
      t = fmaxf(t, __shfl_xor(t, 4, 32));
      t = fmaxf(t, __shfl_xor(t, 8, 32));
      const float mn = fmaxf(rm[r], t);
      const float al = exp2f((rm[r] - mn) * kLog2e);
      const float p0 = exp2f((a - mn) * kLog2e);
      const float p1 = exp2f((c - mn) * kLog2e);
      float ps = p0 + p1;
      ps += __shfl_xor(ps, 1, 32);
      ps += __shfl_xor(ps, 2, 32);
      ps += __shfl_xor(ps, 4, 32);
      ps += __shfl_xor(ps, 8, 32);
      rs[r] = rs[r] * al + ps;
      rm[r] = mn;
      acc[0][r] *= al; acc[1][r] *= al; acc[2][r] *= al; acc[3][r] *= al;
      P[(8 * half + r) * 32 + n]      = (h16)p0;
      P[(8 * half + r) * 32 + 16 + n] = (h16)p1;
    }
    __syncthreads();
    const v16h pa = loadA(P, 32, 0, m, half);
    __syncthreads();
#pragma unroll
    for (int j = 0; j < 4; ++j) {
      v16h vb = loadB(Vbh + (size_t)(16 * j + n) * kS + sk0 + 16 * half);
      acc[j] = wmma_f16(pa, vb, acc[j]);
    }
  }

#pragma unroll
  for (int j = 0; j < 4; ++j)
#pragma unroll
    for (int r = 0; r < 8; ++r) {
      const int s = qt * 16 + 8 * half + r;
      heads[((size_t)(b * kS + s)) * kD + h * kE + 16 * j + n] =
          (h16)(acc[j][r] / rs[r]);
    }
}

// ---- Kernel 5: output projection ------------------------------------------
// out[BS x D] = heads(f16) @ Wo + bo  (Wo pre-transposed to [Dout,Din])
__global__ void out_proj_kernel(const h16* __restrict__ heads,
                                const h16* __restrict__ Wot,
                                const float* __restrict__ bo,
                                float* __restrict__ out) {
  const int lane = threadIdx.x & 31;
  const int wid = blockIdx.x * 4 + (threadIdx.x >> 5);
  const int rt = wid & 255;             // BS/16 = 256 row tiles
  const int ct = wid >> 8;              // D/16  = 64 col tiles
  const int n = lane & 15, m = n, half = lane >> 4;

  const h16* Atile = heads + (size_t)rt * 16 * kD;
  v8f acc = zero8();

  for (int kc = 0; kc < kD / 32; ++kc) {
    v16h a = loadA(Atile, kD, kc * 32, m, half);
    v16h bf = loadB(Wot + (size_t)(ct * 16 + n) * kD + kc * 32 + 16 * half);
    acc = wmma_f16(a, bf, acc);
  }

  const float bias = bo[ct * 16 + n];
#pragma unroll
  for (int r = 0; r < 8; ++r) {
    out[((size_t)(rt * 16 + 8 * half + r)) * kD + ct * 16 + n] = acc[r] + bias;
  }
}

// ---------------------------------------------------------------------------
extern "C" void kernel_launch(void* const* d_in, const int* in_sizes, int n_in,
                              void* d_out, int out_size, void* d_ws,
                              size_t ws_size, hipStream_t stream) {
  const float* query = (const float*)d_in[0];
  const float* value = (const float*)d_in[1];
  const float* Wq = (const float*)d_in[2];
  const float* bq = (const float*)d_in[3];
  const float* Wk = (const float*)d_in[4];
  const float* bk = (const float*)d_in[5];
  const float* Wv = (const float*)d_in[6];
  const float* bv = (const float*)d_in[7];
  const float* Wo = (const float*)d_in[8];
  const float* bo = (const float*)d_in[9];
  float* out = (float*)d_out;

  // workspace layout (bytes); total = 56 MB
  constexpr size_t SZ_ACT = (size_t)kB * kS * kD * sizeof(h16); // 8 MB
  constexpr size_t SZ_W   = (size_t)kH * kD * kE * sizeof(h16); // 2 MB
  constexpr size_t SZ_WO  = (size_t)kD * kD * sizeof(h16);      // 2 MB
  constexpr size_t SZ_QKV = (size_t)kB * kH * kS * kE * sizeof(h16); // 8 MB

  char* ws = (char*)d_ws;
  h16* q16   = (h16*)(ws);
  h16* v16   = (h16*)(ws + SZ_ACT);
  h16* Wqt   = (h16*)(ws + 2 * SZ_ACT);
  h16* Wkt   = (h16*)(ws + 2 * SZ_ACT + SZ_W);
  h16* Wvt   = (h16*)(ws + 2 * SZ_ACT + 2 * SZ_W);
  h16* Wot   = (h16*)(ws + 2 * SZ_ACT + 3 * SZ_W);
  h16* Qb    = (h16*)(ws + 2 * SZ_ACT + 3 * SZ_W + SZ_WO);
  h16* Kb    = (h16*)(ws + 2 * SZ_ACT + 3 * SZ_W + SZ_WO + SZ_QKV);
  h16* Vt    = (h16*)(ws + 2 * SZ_ACT + 3 * SZ_W + SZ_WO + 2 * SZ_QKV);
  h16* heads = (h16*)(ws + 2 * SZ_ACT + 3 * SZ_W + SZ_WO + 3 * SZ_QKV);

  const int nAct = kB * kS * kD;
  cast_acts_kernel<<<4096, 256, 0, stream>>>(query, value, q16, v16, nAct);
  prep_weights_kernel<<<4096, 256, 0, stream>>>(Wq, Wk, Wv, Wo,
                                                Wqt, Wkt, Wvt, Wot);
  // 3 tensors * B * H * (S/16) = 12288 waves, 4 waves/block
  qkv_proj_kernel<<<3072, 128, 0, stream>>>(q16, v16, Wqt, Wkt, Wvt,
                                            bq, bk, bv, Qb, Kb, Vt);
  // B * H * (S/16) = 4096 waves
  flash_attn_kernel<<<1024, 128, 0, stream>>>(Qb, Kb, Vt, heads);
  // (BS/16) * (D/16) = 16384 waves
  out_proj_kernel<<<4096, 128, 0, stream>>>(heads, Wot, bo, out);
}
